// SubgraphSamplingGNNClassifier_30777735643465
// MI455X (gfx1250) — compile-verified, matrix operand order checked by reading, not run on hardware
//
#include <hip/hip_runtime.h>
#include <hip/hip_bf16.h>
#include <cstdint>
#include <cstddef>

typedef __attribute__((ext_vector_type(16))) __bf16 v16bf;
typedef __attribute__((ext_vector_type(8)))  __bf16 v8bf;
typedef __attribute__((ext_vector_type(8)))  float  v8f;

#define S_SUB   4096
#define KNODE   32
#define NSUB    131072   /* S_SUB*KNODE */
#define EPG_    64
#define ESUB    262144   /* S_SUB*EPG_ */
#define HDIM    256
#define INCH    128
#define EDIM    16
#define LAYERS  5
#define GRAPHS  128
#define NCLS    10
#define BN_EPS  1e-5f

// ---------------------------------------------------------------------------
// Weight convert + transpose: src [K][N] f32 row-major -> dst [N][K] bf16
// so that B fragments are contiguous 8-element (16 B) runs in K.
// ---------------------------------------------------------------------------
__global__ __launch_bounds__(256)
void convert_transpose_bf16(const float* __restrict__ src, __bf16* __restrict__ dst,
                            int K, int N) {
    int n = blockIdx.x;
    for (int k = threadIdx.x; k < K; k += blockDim.x)
        dst[(size_t)n * K + k] = (__bf16)src[(size_t)k * N + n];
}

// ---------------------------------------------------------------------------
// WMMA GEMM: C[M,N] (bf16) = gather(A) @ B + bias, optional relu.
//   A_BF16=0: A is f32 (x_global path, with row gather)
//   A_BF16=1: A is bf16 activations
//   BT: bf16 [N][K] transposed weights.
// Block = 256 threads (8 waves), tile = 64 rows x 256 cols.
//   wave (wy 0..3, wx 0..1): 16 rows x 128 cols = 8 WMMA 16x16 tiles.
// Fragment layouts per CDNA5 ISA 7.12.2:
//   16-bit A 16x32: lane(r,g); VGPR v0-3 -> k = g*8 + 0..7 (contiguous),
//                   v4-7 -> k = 16 + g*8 + 0..7 (contiguous).
//   f32 C/D 16x16:  VGPR i -> row = i + 8*g, col = r.
// ---------------------------------------------------------------------------
template <int A_BF16>
__global__ __launch_bounds__(256)
void gemm_bf16_wmma(const void* __restrict__ Av, const int* __restrict__ rowidx,
                    const __bf16* __restrict__ BT, const float* __restrict__ bias,
                    __bf16* __restrict__ C, int N, int K, int lda, int relu) {
    const int lane = threadIdx.x & 31;
    const int wave = threadIdx.x >> 5;
    const int wy   = wave >> 1;       // 0..3 -> 16-row slice
    const int wx   = wave & 1;        // 0..1 -> 128-col half
    const int r    = lane & 15;       // row (A) / col (B) within tile
    const int g    = lane >> 4;       // K-half select

    const int row0 = blockIdx.x * 64 + wy * 16;
    const int arow = row0 + r;
    const int asrc = rowidx ? rowidx[arow] : arow;

    const float*  Arf = (const float*)Av  + (size_t)asrc * lda;
    const __bf16* Arb = (const __bf16*)Av + (size_t)asrc * lda;

    v8f acc[8] = {};

    for (int kk = 0; kk < K; kk += 32) {
        // ---- A fragment ----
        v16bf af;
        if (A_BF16) {
            v8bf lo = *(const v8bf*)(Arb + kk + g * 8);
            v8bf hi = *(const v8bf*)(Arb + kk + 16 + g * 8);
            af = __builtin_shufflevector(lo, hi, 0, 1, 2, 3, 4, 5, 6, 7,
                                                 8, 9, 10, 11, 12, 13, 14, 15);
            if (kk + 32 < K)
                __builtin_prefetch(Arb + kk + 32 + g * 8, 0, 1);
        } else {
            #pragma unroll
            for (int v = 0; v < 8; ++v) {
                int k = kk + ((v >= 4) ? 16 : 0) + g * 8 + (v & 3) * 2;
                af[2 * v]     = (__bf16)Arf[k];
                af[2 * v + 1] = (__bf16)Arf[k + 1];
            }
            if (kk + 32 < K)
                __builtin_prefetch(Arf + kk + 32 + g * 8, 0, 1);
        }
        // ---- 8 column tiles ----
        #pragma unroll
        for (int nt = 0; nt < 8; ++nt) {
            int n0 = wx * 128 + nt * 16;
            const __bf16* Bp = BT + (size_t)(n0 + r) * K + kk + g * 8;
            v8bf blo = *(const v8bf*)(Bp);
            v8bf bhi = *(const v8bf*)(Bp + 16);
            v16bf bf = __builtin_shufflevector(blo, bhi, 0, 1, 2, 3, 4, 5, 6, 7,
                                                         8, 9, 10, 11, 12, 13, 14, 15);
            acc[nt] = __builtin_amdgcn_wmma_f32_16x16x32_bf16(
                false, af, false, bf, (short)0, acc[nt], false, false);
        }
    }

    // ---- epilogue: bias (+relu), store bf16 ----
    #pragma unroll
    for (int nt = 0; nt < 8; ++nt) {
        int col = wx * 128 + nt * 16 + r;
        float b = bias ? bias[col] : 0.0f;
        #pragma unroll
        for (int i = 0; i < 8; ++i) {
            int row = row0 + i + g * 8;
            float val = acc[nt][i] + b;
            if (relu) val = fmaxf(val, 0.0f);
            C[(size_t)row * N + col] = (__bf16)val;
        }
    }
}

// ---------------------------------------------------------------------------
// Edge projection: e = edge_attr[esrc] @ We + be   (K=16, memory-bound)
// ---------------------------------------------------------------------------
__global__ __launch_bounds__(256)
void edge_project(const float* __restrict__ edge_attr, const int* __restrict__ esrc,
                  const float* __restrict__ We, const float* __restrict__ be,
                  __bf16* __restrict__ e) {
    int eid = blockIdx.x;
    int c = threadIdx.x;
    __shared__ float a[EDIM];
    if (c < EDIM) a[c] = edge_attr[(size_t)esrc[eid] * EDIM + c];
    __syncthreads();
    float v = be[c];
    #pragma unroll
    for (int k = 0; k < EDIM; ++k) v = fmaf(a[k], We[k * HDIM + c], v);
    e[(size_t)eid * HDIM + c] = (__bf16)v;
}

// ---------------------------------------------------------------------------
// GINE aggregation: one block per subgraph. Thread c owns channel c for all
// 32 destination nodes; agg tile lives in LDS (f32) -> no atomics at all.
// z = (1+eps)*h + segment_sum(relu(h[src] + e), dst)
// ---------------------------------------------------------------------------
__global__ __launch_bounds__(256)
void gine_aggregate(const __bf16* __restrict__ h, const __bf16* __restrict__ e,
                    const int* __restrict__ ei_src, const int* __restrict__ ei_dst,
                    const int* __restrict__ edge_ptr, const float* __restrict__ eps,
                    int layer, __bf16* __restrict__ z) {
    int s = blockIdx.x;
    int c = threadIdx.x;
    __shared__ float lagg[KNODE * HDIM];   // 32 KB
    __shared__ int ls[EPG_], ld[EPG_];

    #pragma unroll
    for (int k = 0; k < KNODE; ++k) lagg[k * HDIM + c] = 0.0f;

    int e0 = edge_ptr[s];
    int ne = edge_ptr[s + 1] - e0;
    if (ne > EPG_) ne = EPG_;
    for (int j = threadIdx.x; j < ne; j += blockDim.x) {
        ls[j] = ei_src[e0 + j];
        ld[j] = ei_dst[e0 + j];
    }
    __syncthreads();

    const size_t base = (size_t)s * KNODE;
    for (int j = 0; j < ne; ++j) {
        float m = (float)h[(base + ls[j]) * HDIM + c] +
                  (float)e[(size_t)(e0 + j) * HDIM + c];
        m = fmaxf(m, 0.0f);
        lagg[ld[j] * HDIM + c] += m;     // thread-private (dst, c) slot
    }
    __syncthreads();

    float ep = 1.0f + eps[layer];
    #pragma unroll
    for (int k = 0; k < KNODE; ++k) {
        size_t idx = (base + k) * HDIM + c;
        z[idx] = (__bf16)(ep * (float)h[idx] + lagg[k * HDIM + c]);
    }
}

// ---------------------------------------------------------------------------
// BatchNorm: f32 stats over bf16 activations
// ---------------------------------------------------------------------------
__global__ __launch_bounds__(512)
void zero_f32(float* __restrict__ p, int n) {
    int i = blockIdx.x * blockDim.x + threadIdx.x;
    if (i < n) p[i] = 0.0f;
}

__global__ __launch_bounds__(256)
void bn_stats(const __bf16* __restrict__ zv, float* __restrict__ sum,
              float* __restrict__ sumsq, int rows) {
    int c = threadIdx.x;
    float s = 0.0f, q = 0.0f;
    for (int r = blockIdx.x; r < rows; r += gridDim.x) {
        float v = (float)zv[(size_t)r * HDIM + c];
        s += v;
        q = fmaf(v, v, q);
    }
    atomicAdd(&sum[c], s);
    atomicAdd(&sumsq[c], q);
}

// h = relu((z-mu)*rsqrt(var+eps)*gamma + beta) + h   (residual in place)
__global__ __launch_bounds__(256)
void bn_apply_relu_res(const __bf16* __restrict__ zv, const float* __restrict__ sum,
                       const float* __restrict__ sumsq, const float* __restrict__ gamma,
                       const float* __restrict__ beta, __bf16* __restrict__ h, int rows) {
    size_t i = (size_t)blockIdx.x * blockDim.x + threadIdx.x;
    int c = threadIdx.x;   // blockDim == HDIM
    float inv_n = 1.0f / (float)rows;
    float mu  = sum[c] * inv_n;
    float var = sumsq[c] * inv_n - mu * mu;
    float inv = rsqrtf(var + BN_EPS);
    float v = ((float)zv[i] - mu) * inv * gamma[c] + beta[c];
    h[i] = (__bf16)(fmaxf(v, 0.0f) + (float)h[i]);
}

// ---------------------------------------------------------------------------
// Hierarchical mean pool: g[gi] = mean over subgraphs of mean over K nodes
// ---------------------------------------------------------------------------
__global__ __launch_bounds__(256)
void pool_graphs(const __bf16* __restrict__ h, const int* __restrict__ sample_ptr,
                 float* __restrict__ gfeat) {
    int gi = blockIdx.x;
    int c = threadIdx.x;
    int s0 = sample_ptr[gi], s1 = sample_ptr[gi + 1];
    float acc = 0.0f;
    for (int s = s0; s < s1; ++s) {
        size_t base = (size_t)s * KNODE;
        #pragma unroll
        for (int k = 0; k < KNODE; ++k) acc += (float)h[(base + k) * HDIM + c];
    }
    int cnt = (s1 - s0) > 0 ? (s1 - s0) : 1;
    gfeat[(size_t)gi * HDIM + c] = acc / ((float)KNODE * (float)cnt);
}

// ---------------------------------------------------------------------------
// Classifier: out = relu(g @ W1 + b1) @ W2 + b2   (128x256 @ 256x256 @ 256x10)
// ---------------------------------------------------------------------------
__global__ __launch_bounds__(256)
void classifier(const float* __restrict__ g, const float* __restrict__ W1,
                const float* __restrict__ b1, const float* __restrict__ W2,
                const float* __restrict__ b2, float* __restrict__ out) {
    int gi = blockIdx.x;
    int c = threadIdx.x;
    __shared__ float row[HDIM], hid[HDIM];
    row[c] = g[(size_t)gi * HDIM + c];
    __syncthreads();
    float a = b1[c];
    for (int k = 0; k < HDIM; ++k) a = fmaf(row[k], W1[k * HDIM + c], a);
    hid[c] = fmaxf(a, 0.0f);
    __syncthreads();
    if (c < NCLS) {
        float o = b2[c];
        for (int k = 0; k < HDIM; ++k) o = fmaf(hid[k], W2[k * NCLS + c], o);
        out[gi * NCLS + c] = o;
    }
}

// ---------------------------------------------------------------------------
extern "C" void kernel_launch(void* const* d_in, const int* in_sizes, int n_in,
                              void* d_out, int out_size, void* d_ws, size_t ws_size,
                              hipStream_t stream) {
    const float* x_global  = (const float*)d_in[0];
    const float* edge_attr = (const float*)d_in[1];
    const float* Wn        = (const float*)d_in[2];
    const float* bn_       = (const float*)d_in[3];
    const float* We        = (const float*)d_in[4];
    const float* be        = (const float*)d_in[5];
    const float* eps       = (const float*)d_in[6];
    const float* convW1    = (const float*)d_in[7];
    const float* convB1    = (const float*)d_in[8];
    const float* convW2    = (const float*)d_in[9];
    const float* convB2    = (const float*)d_in[10];
    const float* gamma     = (const float*)d_in[11];
    const float* beta      = (const float*)d_in[12];
    const float* clfW1     = (const float*)d_in[13];
    const float* clfB1     = (const float*)d_in[14];
    const float* clfW2     = (const float*)d_in[15];
    const float* clfB2     = (const float*)d_in[16];
    const int* nodes       = (const int*)d_in[17];
    const int* edge_index  = (const int*)d_in[18];
    const int* edge_ptr    = (const int*)d_in[19];
    const int* sample_ptr  = (const int*)d_in[20];
    const int* esrc_g      = (const int*)d_in[21];

    const int* ei_src = edge_index;
    const int* ei_dst = edge_index + ESUB;

    // ---- workspace carve-out ----
    char* ws = (char*)d_ws;
    size_t off = 0;
    auto alloc = [&](size_t bytes) -> void* {
        void* p = (void*)(ws + off);
        off += (bytes + 255) & ~(size_t)255;
        return p;
    };
    __bf16* h     = (__bf16*)alloc((size_t)NSUB * HDIM * sizeof(__bf16));
    __bf16* z     = (__bf16*)alloc((size_t)NSUB * HDIM * sizeof(__bf16));  // also z3
    __bf16* t1    = (__bf16*)alloc((size_t)NSUB * HDIM * sizeof(__bf16));
    __bf16* ef    = (__bf16*)alloc((size_t)ESUB * HDIM * sizeof(__bf16));
    __bf16* WnT   = (__bf16*)alloc((size_t)HDIM * INCH * sizeof(__bf16));
    __bf16* W1T   = (__bf16*)alloc((size_t)LAYERS * HDIM * HDIM * sizeof(__bf16));
    __bf16* W2T   = (__bf16*)alloc((size_t)LAYERS * HDIM * HDIM * sizeof(__bf16));
    float*  bsum  = (float*)alloc((size_t)HDIM * sizeof(float));
    float*  bsq   = (float*)alloc((size_t)HDIM * sizeof(float));
    float*  gfeat = (float*)alloc((size_t)GRAPHS * HDIM * sizeof(float));
    (void)ws_size; (void)in_sizes; (void)n_in; (void)out_size;

    // ---- 1. weights -> bf16 transposed ----
    convert_transpose_bf16<<<HDIM, 256, 0, stream>>>(Wn, WnT, INCH, HDIM);
    for (int l = 0; l < LAYERS; ++l) {
        convert_transpose_bf16<<<HDIM, 256, 0, stream>>>(
            convW1 + (size_t)l * HDIM * HDIM, W1T + (size_t)l * HDIM * HDIM, HDIM, HDIM);
        convert_transpose_bf16<<<HDIM, 256, 0, stream>>>(
            convW2 + (size_t)l * HDIM * HDIM, W2T + (size_t)l * HDIM * HDIM, HDIM, HDIM);
    }

    // ---- 2. edge features ----
    edge_project<<<ESUB, 256, 0, stream>>>(edge_attr, esrc_g, We, be, ef);

    // ---- 3. node projection (fused gather): h = x_global[nodes] @ Wn + bn ----
    gemm_bf16_wmma<0><<<NSUB / 64, 256, 0, stream>>>(
        (const void*)x_global, nodes, WnT, bn_, h, HDIM, INCH, INCH, 0);

    // ---- 4. GINE layers ----
    for (int l = 0; l < LAYERS; ++l) {
        gine_aggregate<<<S_SUB, 256, 0, stream>>>(
            h, ef, ei_src, ei_dst, edge_ptr, eps, l, z);

        gemm_bf16_wmma<1><<<NSUB / 64, 256, 0, stream>>>(
            (const void*)z, nullptr, W1T + (size_t)l * HDIM * HDIM, convB1 + l * HDIM,
            t1, HDIM, HDIM, HDIM, 1);

        gemm_bf16_wmma<1><<<NSUB / 64, 256, 0, stream>>>(
            (const void*)t1, nullptr, W2T + (size_t)l * HDIM * HDIM, convB2 + l * HDIM,
            z, HDIM, HDIM, HDIM, 0);   // z now holds pre-BN z3

        zero_f32<<<1, 512, 0, stream>>>(bsum, HDIM * 2);  // bsum+bsq contiguous
        bn_stats<<<512, 256, 0, stream>>>(z, bsum, bsq, NSUB);
        bn_apply_relu_res<<<NSUB, 256, 0, stream>>>(
            z, bsum, bsq, gamma + l * HDIM, beta + l * HDIM, h, NSUB);
    }

    // ---- 5. pooling + classifier ----
    pool_graphs<<<GRAPHS, 256, 0, stream>>>(h, sample_ptr, gfeat);
    classifier<<<GRAPHS, 256, 0, stream>>>(
        gfeat, clfW1, clfB1, clfW2, clfB2, (float*)d_out);
}
